// SelfAttention_44444321579194
// MI455X (gfx1250) — matrix operand order repaired and checked by reading, not compiled, not run on hardware
//
#include <hip/hip_runtime.h>
#include <hip/hip_bf16.h>

#define BD 2
#define LD 384
#define HIDD 512
#define HD 8
#define DD 64

typedef __attribute__((ext_vector_type(8)))  float          v8f;
typedef __attribute__((ext_vector_type(16))) __bf16         v16bf;
typedef __attribute__((ext_vector_type(16))) unsigned short v16us;
typedef __attribute__((ext_vector_type(8)))  unsigned short v8us;

__device__ __forceinline__ unsigned short f2bf(float f) {   // scalar path (staging only)
    union { float f; unsigned u; } x; x.f = f;
    unsigned r = x.u + 0x7FFFu + ((x.u >> 16) & 1u);
    return (unsigned short)(r >> 16);
}

// A-fragment (16x32 bf16, wave32): lane<16 row=lane K={0..15}, lane>=16 K={16..31}.
// rowptr already includes row*lda + kc*32 + ((lane>>4)*16). Two aligned 32B fp32 loads,
// converted with packed v_cvt_pk_bf16_f32.
__device__ __forceinline__ v16bf load_afrag_f32(const float* rowptr) {
    v8f a0 = *(const v8f*)(rowptr);
    v8f a1 = *(const v8f*)(rowptr + 8);
    v16bf r;
#pragma unroll
    for (int i = 0; i < 8; ++i) { r[i] = (__bf16)a0[i]; r[8 + i] = (__bf16)a1[i]; }
    return r;
}

// B-fragment: 16 contiguous bf16 (pre-fragified global or transposed LDS).
__device__ __forceinline__ v16bf load_bfrag(const unsigned short* p) {
    v8us lo = *(const v8us*)(p);
    v8us hi = *(const v8us*)(p + 8);
    v16us u;
#pragma unroll
    for (int i = 0; i < 8; ++i) { u[i] = lo[i]; u[8 + i] = hi[i]; }
    return __builtin_bit_cast(v16bf, u);
}

__device__ __forceinline__ v8f wmma_bf16(v16bf a, v16bf b, v8f c) {
    return __builtin_amdgcn_wmma_f32_16x16x32_bf16(false, a, false, b, (short)0, c, false, false);
}

__device__ __forceinline__ v8f vzero8() { return v8f{0.f,0.f,0.f,0.f,0.f,0.f,0.f,0.f}; }

// ---------------- K0: fragify weights into per-lane WMMA-B order (bf16) ----------------
// dst[((kc*NT + nt)*32 + lane)*16 + e] = bf16(src[k*N + n]),
//   k = kc*32 + (lane>>4)*16 + e ; n = nt*16 + (lane&15)
__global__ void fragify_w(const float* __restrict__ src, unsigned short* __restrict__ dst,
                          int K, int N) {
    int idx = blockIdx.x * 256 + threadIdx.x;
    if (idx >= K * N) return;
    int e    = idx & 15;
    int lane = (idx >> 4) & 31;
    int fr   = idx >> 9;
    int NT   = N >> 4;
    int kc   = fr / NT, nt = fr % NT;
    int k = kc * 32 + ((lane >> 4) << 4) + e;
    int n = nt * 16 + (lane & 15);
    dst[idx] = f2bf(src[(size_t)k * N + n]);
}

// ---------------- K1: QKV projection, M=768 N=512 K=512, bf16 WMMA ----------------
// 8 waves * unique 16-row tiles (M=128/WG); per k-step: prefetch 8 B-frags, then 8 WMMAs.
__global__ __launch_bounds__(256) void qkv_gemm(
    const float* __restrict__ hidden,
    const unsigned short* __restrict__ wqF, const unsigned short* __restrict__ wkF,
    const unsigned short* __restrict__ wvF,
    const float* __restrict__ bq, const float* __restrict__ bk, const float* __restrict__ bv,
    float* __restrict__ qo, float* __restrict__ ko, float* __restrict__ vo) {
    const unsigned short* wF; const float* bias; float* out;
    if (blockIdx.z == 0)      { wF = wqF; bias = bq; out = qo; }
    else if (blockIdx.z == 1) { wF = wkF; bias = bk; out = ko; }
    else                      { wF = wvF; bias = bv; out = vo; }

    int tid = threadIdx.x, lid = tid & 31, w = tid >> 5;
    int mrow = blockIdx.x * 128 + w * 16;
    int n0   = blockIdx.y * 128;

    v8f acc[8];
#pragma unroll
    for (int nt = 0; nt < 8; ++nt) acc[nt] = vzero8();

    const float* abase = hidden + (size_t)(mrow + (lid & 15)) * HIDD + ((lid >> 4) * 16);
#pragma unroll 2
    for (int kc = 0; kc < 16; ++kc) {
        v16bf a = load_afrag_f32(abase + kc * 32);
        v16bf bf[8];
#pragma unroll
        for (int nt = 0; nt < 8; ++nt) {
            int ng = (n0 >> 4) + nt;                    // global n-tile, NT=32
            bf[nt] = load_bfrag(wF + ((size_t)(kc * 32 + ng) * 32 + lid) * 16);
        }
#pragma unroll
        for (int nt = 0; nt < 8; ++nt) acc[nt] = wmma_bf16(a, bf[nt], acc[nt]);
    }
    int hi = (lid >> 4) * 8;
#pragma unroll
    for (int nt = 0; nt < 8; ++nt) {
        int col = n0 + nt * 16 + (lid & 15);
        float bb = bias[col];
#pragma unroll
        for (int r = 0; r < 8; ++r)
            out[(size_t)(mrow + r + hi) * HIDD + col] = acc[nt][r] + bb;
    }
}

// ---------------- K2: fused structure projection + score assembly ----------------
// Chunks of 128 r-rows; each of 8 waves owns a unique 16-row tile and computes BOTH
// sq (4 n-tiles) and sk (4 n-tiles): 1 A-frag conversion feeds 8 WMMAs.
__global__ __launch_bounds__(256) void struct_scores(
    const float* __restrict__ sm, const float* __restrict__ qb, const float* __restrict__ kb,
    const unsigned short* __restrict__ wsqF, const unsigned short* __restrict__ wskF,
    const float* __restrict__ bsq, const float* __restrict__ bsk,
    const float* __restrict__ mask, float* __restrict__ scores) {
    __shared__ float sS[2][128][68];                    // [sq|sk][r_local][d], padded

    int l = blockIdx.x, b = blockIdx.y;
    int tid = threadIdx.x, lid = tid & 31, w = tid >> 5;
    int mrl = w * 16;                                   // this wave's row tile in chunk
    const size_t smrow0 = (size_t)(b * LD + l) * LD;

    for (int rc = 0; rc < 3; ++rc) {
        int r0 = rc * 128;
        v8f accQ[4], accK[4];
#pragma unroll
        for (int nt = 0; nt < 4; ++nt) { accQ[nt] = vzero8(); accK[nt] = vzero8(); }

        const float* abase = sm + (smrow0 + r0 + mrl + (lid & 15)) * HIDD + ((lid >> 4) * 16);
#pragma unroll 2
        for (int kc = 0; kc < 16; ++kc) {
            v16bf a = load_afrag_f32(abase + kc * 32);
            v16bf fq[4], fk[4];
#pragma unroll
            for (int nt = 0; nt < 4; ++nt) {            // NT = 4 (D=64)
                size_t foff = ((size_t)(kc * 4 + nt) * 32 + lid) * 16;
                fq[nt] = load_bfrag(wsqF + foff);
                fk[nt] = load_bfrag(wskF + foff);
            }
#pragma unroll
            for (int nt = 0; nt < 4; ++nt) {
                accQ[nt] = wmma_bf16(a, fq[nt], accQ[nt]);
                accK[nt] = wmma_bf16(a, fk[nt], accK[nt]);
            }
        }
        int hi = (lid >> 4) * 8;
#pragma unroll
        for (int nt = 0; nt < 4; ++nt) {
            int col = nt * 16 + (lid & 15);
            float bq_ = bsq[col], bk_ = bsk[col];
#pragma unroll
            for (int r = 0; r < 8; ++r) {
                sS[0][mrl + r + hi][col] = accQ[nt][r] + bq_;
                sS[1][mrl + r + hi][col] = accK[nt][r] + bk_;
            }
        }
        __syncthreads();

        // score[h, r] = ( q_l . (k_r + sq_r) + k_r . sk_r ) / 4 + mask[r]
        int h  = w;                                     // 8 waves == 8 heads
        int rl = tid & 31;
        const float4* q4 = (const float4*)(qb + (size_t)(b * LD + l) * HIDD + h * DD);
#pragma unroll
        for (int i = 0; i < 4; ++i) {
            int rloc = rl + i * 32;
            int rg   = r0 + rloc;
            const float4* k4 = (const float4*)(kb + (size_t)(b * LD + rg) * HIDD + h * DD);
            const float4* s4 = (const float4*)(&sS[0][rloc][0]);
            const float4* t4 = (const float4*)(&sS[1][rloc][0]);
            float s = 0.f;
#pragma unroll
            for (int j = 0; j < 16; ++j) {
                float4 qv = q4[j], kv = k4[j], sv = s4[j], tv = t4[j];
                s += qv.x * (kv.x + sv.x) + kv.x * tv.x;
                s += qv.y * (kv.y + sv.y) + kv.y * tv.y;
                s += qv.z * (kv.z + sv.z) + kv.z * tv.z;
                s += qv.w * (kv.w + sv.w) + kv.w * tv.w;
            }
            s = s * 0.25f + mask[b * LD + rg];
            scores[((size_t)(b * HD + h) * LD + l) * LD + rg] = s;
        }
        __syncthreads();
    }
}

// ---------------- K3: row softmax in place (rows of 384) ----------------
__global__ __launch_bounds__(128) void softmax_rows(float* __restrict__ scores) {
    __shared__ float red[128];
    int t = threadIdx.x;
    float* row = scores + (size_t)blockIdx.x * LD;
    float a = row[t], b = row[t + 128], c = row[t + 256];
    float m = fmaxf(a, fmaxf(b, c));
    red[t] = m; __syncthreads();
    for (int s = 64; s > 0; s >>= 1) { if (t < s) red[t] = fmaxf(red[t], red[t + s]); __syncthreads(); }
    float M = red[0]; __syncthreads();
    float e0 = expf(a - M), e1 = expf(b - M), e2 = expf(c - M);
    red[t] = e0 + e1 + e2; __syncthreads();
    for (int s = 64; s > 0; s >>= 1) { if (t < s) red[t] += red[t + s]; __syncthreads(); }
    float inv = 1.0f / red[0];
    row[t] = e0 * inv; row[t + 128] = e1 * inv; row[t + 256] = e2 * inv;
}

// ---------------- K4: ctx = probs @ V, per (b,h), WMMA bf16; out [B,L,H*D] ----------------
__global__ __launch_bounds__(256) void ctx_gemm(
    const float* __restrict__ probs, const float* __restrict__ vb, float* __restrict__ out) {
    __shared__ unsigned short sVT[64][392];             // V^T (d-major), padded
    int bh = blockIdx.y; int b = bh >> 3, h = bh & 7;
    int tid = threadIdx.x, lid = tid & 31, w = tid >> 5;

    for (int i = tid; i < LD * DD; i += 256) {
        int r = i >> 6, d = i & 63;
        sVT[d][r] = f2bf(vb[(size_t)(b * LD + r) * HIDD + h * DD + d]);
    }
    __syncthreads();

    int mrow = blockIdx.x * 128 + w * 16;
    v8f acc[4];
#pragma unroll
    for (int nt = 0; nt < 4; ++nt) acc[nt] = vzero8();

    const float* abase = probs + ((size_t)bh * LD + mrow + (lid & 15)) * LD + ((lid >> 4) * 16);
#pragma unroll 2
    for (int kc = 0; kc < 12; ++kc) {
        v16bf a = load_afrag_f32(abase + kc * 32);
        v16bf bf[4];
#pragma unroll
        for (int nt = 0; nt < 4; ++nt)
            bf[nt] = load_bfrag(&sVT[nt * 16 + (lid & 15)][kc * 32 + ((lid >> 4) * 16)]);
#pragma unroll
        for (int nt = 0; nt < 4; ++nt) acc[nt] = wmma_bf16(a, bf[nt], acc[nt]);
    }
    int hi = (lid >> 4) * 8;
#pragma unroll
    for (int nt = 0; nt < 4; ++nt) {
        int col = nt * 16 + (lid & 15);
#pragma unroll
        for (int r = 0; r < 8; ++r) {
            int row = mrow + r + hi;
            out[(size_t)(b * LD + row) * HIDD + h * DD + col] = acc[nt][r];
        }
    }
}

extern "C" void kernel_launch(void* const* d_in, const int* in_sizes, int n_in,
                              void* d_out, int out_size, void* d_ws, size_t ws_size,
                              hipStream_t stream) {
    (void)in_sizes; (void)n_in; (void)out_size; (void)ws_size;
    const float* hidden = (const float*)d_in[0];
    const float* smx    = (const float*)d_in[1];
    const float* mask   = (const float*)d_in[2];
    const float* Wq = (const float*)d_in[3];  const float* bq = (const float*)d_in[4];
    const float* Wk = (const float*)d_in[5];  const float* bk = (const float*)d_in[6];
    const float* Wv = (const float*)d_in[7];  const float* bv = (const float*)d_in[8];
    const float* Wsq = (const float*)d_in[9]; const float* bsq = (const float*)d_in[10];
    const float* Wsk = (const float*)d_in[11];const float* bsk = (const float*)d_in[12];
    float* out = (float*)d_out;

    // workspace layout
    float* q      = (float*)d_ws;                 // 393216 f
    float* k      = q + 393216;
    float* v      = k + 393216;
    float* scores = v + 393216;                   // 2359296 f
    unsigned short* wqF  = (unsigned short*)(scores + 2359296); // 262144 halves each
    unsigned short* wkF  = wqF + 262144;
    unsigned short* wvF  = wkF + 262144;
    unsigned short* wsqF = wvF + 262144;          // 32768 halves each
    unsigned short* wskF = wsqF + 32768;

    fragify_w<<<1024, 256, 0, stream>>>(Wq,  wqF,  HIDD, HIDD);
    fragify_w<<<1024, 256, 0, stream>>>(Wk,  wkF,  HIDD, HIDD);
    fragify_w<<<1024, 256, 0, stream>>>(Wv,  wvF,  HIDD, HIDD);
    fragify_w<<<128,  256, 0, stream>>>(Wsq, wsqF, HIDD, DD);
    fragify_w<<<128,  256, 0, stream>>>(Wsk, wskF, HIDD, DD);

    qkv_gemm<<<dim3(6, 4, 3), 256, 0, stream>>>(hidden, wqF, wkF, wvF, bq, bk, bv, q, k, v);
    struct_scores<<<dim3(LD, BD), 256, 0, stream>>>(smx, q, k, wsqF, wskF, bsq, bsk, mask, scores);
    softmax_rows<<<BD * HD * LD, 128, 0, stream>>>(scores);
    ctx_gemm<<<dim3(3, BD * HD), 256, 0, stream>>>(scores, v, out);
}
